// HybridQLSTM_65481071400769
// MI455X (gfx1250) — matrix operand order repaired and checked by reading, not compiled, not run on hardware
//
#include <hip/hip_runtime.h>
#include <hip/hip_bf16.h>

// ---------------------------------------------------------------------------
// HybridQLSTM for MI455X (gfx1250, wave32, WMMA).
//   SEQ=8192, B=1, IN_DIM=4 (+1 conv feature), HID=2048, 4H=8192 gate rows.
//   Phase 0: conv/x-ext precompute, bf16 weight packing into WMMA A/B layouts.
//   Phase 1: persistent recurrent kernel, 64 WGs x 256 thr (512 waves),
//            each wave owns a 16-row gate tile -> v_wmma_f32_16x16x32_bf16
//            matvec (h broadcast across B columns), device-wide barrier per step.
//   Phase 2: logits = hs @ W_cls^T as a real WMMA GEMM over the bf16 history.
// ---------------------------------------------------------------------------

#define SEQ    8192
#define HID    2048
#define IN5    5
#define G4H    (4 * HID)      // 8192 gate rows
#define NWG    64
#define TPB    256
#define WAVES  8              // waves per WG (wave32)
#define HPW    32             // hidden units per WG
#define KC     (HID / 32)     // 64 K-chunks of 32
#define TILES  (NWG * WAVES)  // 512 row-tiles of 16
#define TILE_ELEMS 32768      // KC * 32 lanes * 16 bf16 per tile

typedef __attribute__((ext_vector_type(16))) __bf16 v16bf;
typedef __attribute__((ext_vector_type(8)))  float  v8f;

union BF16x16 { uint4 u[2]; v16bf v; };

__device__ __forceinline__ unsigned short f2bf(float f) {
  unsigned int u = __float_as_uint(f);
  unsigned int r = u + 0x7FFFu + ((u >> 16) & 1u);   // round-nearest-even
  return (unsigned short)(r >> 16);
}
__device__ __forceinline__ float sigmoidf_(float x) {
  return 1.0f / (1.0f + __expf(-x));
}

// --------------------------- phase 0 kernels -------------------------------

__global__ void k_init(const float* __restrict__ h0,
                       unsigned short* __restrict__ hbuf,
                       int* __restrict__ cnt) {
  int gid = blockIdx.x * blockDim.x + threadIdx.x;
  if (gid < HID) {
    unsigned short v = f2bf(h0[gid]);
    hbuf[gid]       = v;     // buffer 0
    hbuf[HID + gid] = v;     // buffer 1
  }
  if (gid == HID) *cnt = 0;  // grid-barrier generation counter
}

__global__ void k_xext(const float* __restrict__ inputs,
                       const float* __restrict__ ck,
                       float* __restrict__ xext) {
  int s = blockIdx.x * blockDim.x + threadIdx.x;
  if (s >= SEQ) return;
  float a0 = inputs[s * 4 + 0], a1 = inputs[s * 4 + 1];
  float a2 = inputs[s * 4 + 2], a3 = inputs[s * 4 + 3];
  float conv = a0 * ck[0] + a1 * ck[1] + a2 * ck[2] + a3 * ck[3];
  xext[s * IN5 + 0] = a0;
  xext[s * IN5 + 1] = a1;
  xext[s * IN5 + 2] = a2;
  xext[s * IN5 + 3] = a3;
  xext[s * IN5 + 4] = sigmoidf_(conv);   // THRESH == 0
}

// Pack W_hh (f32, [4H, HID] row-major) -> bf16 in per-wave-tile WMMA A layout.
// Tile t: WG w=t/8, wave v=t%8; tile row m -> gate g=m>>2, hid j=w*32+v*4+(m&3).
// A 16-bit 16x32 layout: lane l holds row m=l&15; element e -> K offset
// (e<8 ? e : e+8) + (l>=16 ? 8 : 0).
__global__ void k_pack_whh(const float* __restrict__ W_hh,
                           unsigned short* __restrict__ Wp) {
  int gid = blockIdx.x * blockDim.x + threadIdx.x;   // < 4H*HID = 16.7M
  int t   = gid >> 15;            // /32768
  int rem = gid & 32767;
  int c   = rem >> 9;             // K-chunk
  int l   = (rem >> 4) & 31;      // lane
  int e   = rem & 15;             // bf16 element in lane
  int m   = l & 15;
  int k   = c * 32 + (e < 8 ? e : e + 8) + ((l >= 16) ? 8 : 0);
  int g   = m >> 2;
  int j   = (t >> 3) * HPW + (t & 7) * 4 + (m & 3);
  int r   = g * HID + j;
  Wp[gid] = f2bf(W_hh[(size_t)r * HID + k]);
}

// Pack W_cls (f32, [2, HID]) -> bf16 WMMA B fragments per K-chunk.
// B 16-bit 32x16 layout: lane l -> column n=l&15; element e -> K offset
// (e<8 ? e : e+8) + (l>=16 ? 8 : 0).
__global__ void k_pack_wcls(const float* __restrict__ W_cls,
                            unsigned short* __restrict__ Wc) {
  int gid = blockIdx.x * blockDim.x + threadIdx.x;   // < KC*512 = 32768
  int c = gid >> 9;
  int l = (gid >> 4) & 31;
  int e = gid & 15;
  int n = l & 15;
  int k = c * 32 + ((l < 16) ? 0 : 8) + (e < 8 ? e : e + 8);
  Wc[gid] = (n < 2) ? f2bf(W_cls[n * HID + k]) : (unsigned short)0;
}

// --------------------------- phase 1: recurrence ---------------------------

__global__ void __launch_bounds__(TPB) k_lstm(
    const float* __restrict__ xext, const float* __restrict__ W_ih,
    const float* __restrict__ b_ih, const float* __restrict__ b_hh,
    const float* __restrict__ c0,   const unsigned short* __restrict__ Wp,
    unsigned short* hbuf, unsigned short* __restrict__ hs,
    float* __restrict__ out, int* cnt) {
  __shared__ __align__(16) unsigned short h_sh[HID];
  __shared__ float x_sh[8];
  __shared__ float gates_sh[4][HPW];
  __shared__ float c_sh[HPW];

  const int w    = blockIdx.x;
  const int tid  = threadIdx.x;
  const int wave = tid >> 5;
  const int lane = tid & 31;
  const unsigned short* Wt = Wp + (size_t)(w * WAVES + wave) * TILE_ELEMS;
  const int koff = (lane < 16) ? 0 : 8;

  if (tid < HPW) c_sh[tid] = c0[w * HPW + tid];

  int cur = 0;
  for (int s = 0; s < SEQ; ++s) {
    // Broadcast h_t into LDS (256 lanes x 16B = 4KB) + this step's x.
    reinterpret_cast<uint4*>(h_sh)[tid] =
        reinterpret_cast<const uint4*>(hbuf + cur * HID)[tid];
    if (tid < IN5) x_sh[tid] = xext[s * IN5 + tid];
    __syncthreads();

    // gates[16 rows] = W_tile(16x2048) . h(2048) via 64 bf16 WMMAs.
    v8f acc = {0.f, 0.f, 0.f, 0.f, 0.f, 0.f, 0.f, 0.f};
#pragma unroll 4
    for (int c = 0; c < KC; ++c) {
      BF16x16 A, B;
      const uint4* ap = reinterpret_cast<const uint4*>(Wt + c * 512 + lane * 16);
      A.u[0] = ap[0];
      A.u[1] = ap[1];
      const uint4* bp = reinterpret_cast<const uint4*>(&h_sh[c * 32 + koff]);
      B.u[0] = bp[0];
      B.u[1] = bp[2];
      acc = __builtin_amdgcn_wmma_f32_16x16x32_bf16(
          false, A.v, false, B.v, (short)0, acc, false, false);
    }

    // Column 0 of D holds the matvec: lane 0 -> rows 0..7, lane 16 -> 8..15.
    if ((lane & 15) == 0) {
      float x0 = x_sh[0], x1 = x_sh[1], x2 = x_sh[2], x3 = x_sh[3], x4 = x_sh[4];
#pragma unroll
      for (int p = 0; p < 8; ++p) {
        int m  = p + ((lane >> 4) << 3);
        int g  = m >> 2;
        int jl = wave * 4 + (m & 3);
        int r  = g * HID + w * HPW + jl;
        const float* wr = W_ih + (size_t)r * IN5;
        gates_sh[g][jl] = acc[p] + b_ih[r] + b_hh[r] +
                          wr[0] * x0 + wr[1] * x1 + wr[2] * x2 +
                          wr[3] * x3 + wr[4] * x4;
      }
    }
    __syncthreads();

    // LSTM cell pointwise for this WG's 32 hidden units.
    if (tid < HPW) {
      float gi = gates_sh[0][tid], gf = gates_sh[1][tid];
      float gg = gates_sh[2][tid], go = gates_sh[3][tid];
      float cn = sigmoidf_(gf) * c_sh[tid] + sigmoidf_(gi) * tanhf(gg);
      float hn = sigmoidf_(go) * tanhf(cn);
      c_sh[tid] = cn;
      int j = w * HPW + tid;
      unsigned short hb = f2bf(hn);
      hbuf[(cur ^ 1) * HID + j] = hb;        // h_{t+1} for everyone
      hs[(size_t)s * HID + j]   = hb;        // history for the classifier
      if (s == SEQ - 1) {
        out[SEQ * 2 + j]       = hn;         // h_n (f32)
        out[SEQ * 2 + HID + j] = cn;         // c_n (f32)
      }
      __threadfence();
    }
    __syncthreads();

    // Device-wide step barrier: monotonic generation counter.
    if (tid == 0) {
      atomicAdd(cnt, 1);
      int target = (s + 1) * NWG;
      while (__hip_atomic_load(cnt, __ATOMIC_ACQUIRE,
                               __HIP_MEMORY_SCOPE_AGENT) < target)
        __builtin_amdgcn_s_sleep(1);
    }
    __syncthreads();
    cur ^= 1;
  }
}

// --------------------------- phase 2: classifier ---------------------------

__global__ void __launch_bounds__(TPB) k_logits(
    const unsigned short* __restrict__ hs, const unsigned short* __restrict__ Wc,
    const float* __restrict__ b_cls, float* __restrict__ out) {
  const int tid  = threadIdx.x;
  const int wave = tid >> 5;
  const int lane = tid & 31;
  const int t    = blockIdx.x * WAVES + wave;   // 512 tiles of 16 timesteps
  const int m    = lane & 15;
  const int roff = (lane >> 4) << 3;            // K half-offset for A layout

  v8f acc = {0.f, 0.f, 0.f, 0.f, 0.f, 0.f, 0.f, 0.f};
#pragma unroll 4
  for (int c = 0; c < KC; ++c) {
    BF16x16 A, B;
    const unsigned short* ar =
        hs + (size_t)(t * 16 + m) * HID + c * 32 + roff;
    A.u[0] = *reinterpret_cast<const uint4*>(ar);
    A.u[1] = *reinterpret_cast<const uint4*>(ar + 16);
    const uint4* bp = reinterpret_cast<const uint4*>(Wc + c * 512 + lane * 16);
    B.u[0] = bp[0];
    B.u[1] = bp[1];
    acc = __builtin_amdgcn_wmma_f32_16x16x32_bf16(
        false, A.v, false, B.v, (short)0, acc, false, false);
  }

  int n = lane & 15;
  if (n < 2) {                                  // only 2 live output columns
    int mbase = (lane >> 4) << 3;
#pragma unroll
    for (int p = 0; p < 8; ++p) {
      int step = t * 16 + mbase + p;
      out[step * 2 + n] = acc[p] + b_cls[n];
    }
  }
}

// ------------------------------- launcher ----------------------------------

extern "C" void kernel_launch(void* const* d_in, const int* in_sizes, int n_in,
                              void* d_out, int out_size, void* d_ws, size_t ws_size,
                              hipStream_t stream) {
  const float* inputs = (const float*)d_in[0];
  const float* h0     = (const float*)d_in[1];
  const float* c0     = (const float*)d_in[2];
  const float* ck     = (const float*)d_in[3];
  const float* W_ih   = (const float*)d_in[4];
  const float* W_hh   = (const float*)d_in[5];
  const float* b_ih   = (const float*)d_in[6];
  const float* b_hh   = (const float*)d_in[7];
  const float* W_cls  = (const float*)d_in[8];
  const float* b_cls  = (const float*)d_in[9];
  float* out = (float*)d_out;

  // Workspace layout (bytes, 256-aligned sections):
  char* ws = (char*)d_ws;
  float*          xext = (float*)(ws + 0);                        // 163840
  int*            cnt  = (int*)(ws + 163840);                     // 4
  unsigned short* hbuf = (unsigned short*)(ws + 164096);          // 8192
  unsigned short* Wc   = (unsigned short*)(ws + 172288);          // 65536
  unsigned short* Wp   = (unsigned short*)(ws + 237824);          // 33554432
  unsigned short* hs   = (unsigned short*)(ws + 33792256);        // 33554432
  if (ws_size < (size_t)67346688) return;                         // need ~64.3 MB

  k_init<<<9, TPB, 0, stream>>>(h0, hbuf, cnt);
  k_xext<<<SEQ / TPB, TPB, 0, stream>>>(inputs, ck, xext);
  k_pack_whh<<<(G4H * HID) / TPB, TPB, 0, stream>>>(W_hh, Wp);
  k_pack_wcls<<<(KC * 512) / TPB, TPB, 0, stream>>>(W_cls, Wc);
  k_lstm<<<NWG, TPB, 0, stream>>>(xext, W_ih, b_ih, b_hh, c0, Wp, hbuf, hs,
                                  out, cnt);
  k_logits<<<TILES / WAVES, TPB, 0, stream>>>(hs, Wc, b_cls, out);
}
